// PointNetSetAbstractionSsg_10874857193494
// MI455X (gfx1250) — matrix operand order repaired and checked by reading, not compiled, not run on hardware
//
#include <hip/hip_runtime.h>

typedef __attribute__((ext_vector_type(16))) _Float16 v16h;
typedef __attribute__((ext_vector_type(8)))  _Float16 v8h;
typedef __attribute__((ext_vector_type(4)))  _Float16 v4h;
typedef __attribute__((ext_vector_type(8)))  float    v8f;

#define NPTS   4096
#define NGRP   1024
#define KGRP   64
#define NBATCH 16
#define MTOT   (1u<<20)          // 16*1024*64 columns
#define NTILES (MTOT/16)         // 65536 m-tiles
#define R2     0.04f

// ---------------- helpers ----------------

// 16-bit A-matrix (16x32) half->K mapping: contiguous, k = 16*g + h.
__device__ __forceinline__ int kmapA(int h, int g) {
    return g * 16 + h;
}

template<int CIN>
__device__ __forceinline__ v16h load_wfrag(const float* __restrict__ W,
                                           int cout0, int kbase, int lane) {
    const int row = cout0 + (lane & 15);
    const int g   = lane >> 4;
    v16h a;
#pragma unroll
    for (int h = 0; h < 16; ++h) {
        int k = kbase + kmapA(h, g);
        float w = (k < CIN) ? W[row * CIN + k] : 0.0f;
        a[h] = (_Float16)w;
    }
    return a;
}

// ---------------- FPS: 1 block per batch ----------------
__global__ void fps_kernel(const float* __restrict__ coord,
                           float* __restrict__ nxyz,
                           float* __restrict__ out_coord) {
    const int b    = blockIdx.x;
    const int tid  = threadIdx.x;
    const int lane = tid & 31;
    const int wave = tid >> 5;
    __shared__ float cc[3];
    __shared__ float bd[8];
    __shared__ int   bi[8];

    const float* cb = coord + (size_t)b * 3 * NPTS;
    float px[16], py[16], pz[16], dd[16];
#pragma unroll
    for (int i = 0; i < 16; ++i) {
        int p = tid + i * 256;
        px[i] = cb[p]; py[i] = cb[NPTS + p]; pz[i] = cb[2 * NPTS + p];
        dd[i] = 1e10f;
    }

    int far = 0;
    for (int it = 0; it < NGRP; ++it) {
        if (tid == (far & 255)) {
            int sl = far >> 8;
            cc[0] = px[sl]; cc[1] = py[sl]; cc[2] = pz[sl];
            float* o = nxyz + ((size_t)b * NGRP + it) * 3;
            o[0] = px[sl]; o[1] = py[sl]; o[2] = pz[sl];
        }
        __syncthreads();
        float cx = cc[0], cy = cc[1], cz = cc[2];
        float bmax = -1.0f; int bidx = 0;
#pragma unroll
        for (int i = 0; i < 16; ++i) {
            float dx = px[i] - cx, dy = py[i] - cy, dz = pz[i] - cz;
            float d = dx * dx + dy * dy + dz * dz;
            if (d < dd[i]) dd[i] = d;
            if (dd[i] > bmax) { bmax = dd[i]; bidx = tid + i * 256; }
        }
#pragma unroll
        for (int off = 1; off < 32; off <<= 1) {
            float od = __shfl_xor(bmax, off, 32);
            int   oi = __shfl_xor(bidx, off, 32);
            if (od > bmax || (od == bmax && oi < bidx)) { bmax = od; bidx = oi; }
        }
        if (lane == 0) { bd[wave] = bmax; bi[wave] = bidx; }
        __syncthreads();
        float fb = bd[0]; int fi = bi[0];
#pragma unroll
        for (int w = 1; w < 8; ++w)
            if (bd[w] > fb || (bd[w] == fb && bi[w] < fi)) { fb = bd[w]; fi = bi[w]; }
        far = fi;
    }
    __syncthreads();
    // out_coord[b][c][s] = nxyz[b][s][c]
    for (int idx = tid; idx < 3 * NGRP; idx += 256) {
        int c = idx / NGRP, s = idx % NGRP;
        out_coord[((size_t)b * 3 + c) * NGRP + s] =
            nxyz[((size_t)b * NGRP + s) * 3 + c];
    }
}

// ---------------- ball query + grouping: 1 wave per centroid ----------------
__global__ void group_kernel(const float* __restrict__ coord,
                             const float* __restrict__ feat,
                             const float* __restrict__ nxyz,
                             _Float16* __restrict__ A0) {
    const int gw   = blockIdx.x * 8 + (threadIdx.x >> 5);
    const int lane = threadIdx.x & 31;
    const int b = gw >> 10, s = gw & 1023;
    const float* nb = nxyz + ((size_t)b * NGRP + s) * 3;
    const float cx = nb[0], cy = nb[1], cz = nb[2];
    const float* cb = coord + (size_t)b * 3 * NPTS;
    const float* fb = feat  + (size_t)b * 3 * NPTS;
    const size_t mbase = (size_t)(b * NGRP + s) * KGRP;

    int cnt = 0, firstIdx = -1;
    for (int base = 0; base < NPTS && cnt < KGRP; base += 32) {
        int j = base + lane;
        float dx = cb[j] - cx, dy = cb[NPTS + j] - cy, dz = cb[2 * NPTS + j] - cz;
        float d2 = dx * dx + dy * dy + dz * dz;
        bool in = (d2 <= R2);
        unsigned mask = (unsigned)__ballot(in);
        if (firstIdx < 0 && mask) firstIdx = base + __builtin_ctz(mask);
        int pos = cnt + __popc(mask & ((1u << lane) - 1u));
        if (in && pos < KGRP) {
            v8h r;
            r[0] = (_Float16)dx; r[1] = (_Float16)dy; r[2] = (_Float16)dz;
            r[3] = (_Float16)fb[j]; r[4] = (_Float16)fb[NPTS + j]; r[5] = (_Float16)fb[2 * NPTS + j];
            r[6] = (_Float16)0; r[7] = (_Float16)0;
            *(v8h*)(A0 + (mbase + pos) * 8) = r;
        }
        cnt += __popc(mask);
    }
    if (cnt < KGRP && firstIdx >= 0) {
        int j = firstIdx;
        float dx = cb[j] - cx, dy = cb[NPTS + j] - cy, dz = cb[2 * NPTS + j] - cz;
        v8h r;
        r[0] = (_Float16)dx; r[1] = (_Float16)dy; r[2] = (_Float16)dz;
        r[3] = (_Float16)fb[j]; r[4] = (_Float16)fb[NPTS + j]; r[5] = (_Float16)fb[2 * NPTS + j];
        r[6] = (_Float16)0; r[7] = (_Float16)0;
        for (int q = cnt + lane; q < KGRP; q += 32)
            *(v8h*)(A0 + (mbase + q) * 8) = r;
    }
}

// ---------------- fused GEMM + (input norm/relu) + stats ----------------
// Y[m][COUT] = W[COUT][CIN] * act(Ain[m][CS]); raw f16 output + channel stats.
template<int CIN, int CS, int COUT, bool FIRST>
__global__ void gemm_kernel(const _Float16* __restrict__ Ain,
                            const float* __restrict__ W,
                            const _Float16* __restrict__ scale,
                            const _Float16* __restrict__ shift,
                            _Float16* __restrict__ Yout,
                            float* __restrict__ gsum,
                            float* __restrict__ gsq) {
    constexpr int NCT = COUT / 16;       // cout tiles
    constexpr int MPB = 8 / NCT;         // m-tiles per block per stride step
    constexpr int NK  = (CIN + 31) / 32; // k-steps
    const int lane = threadIdx.x & 31;
    const int wave = threadIdx.x >> 5;
    const int ct    = wave % NCT;
    const int msub  = wave / NCT;
    const int cout0 = ct * 16;
    const int g     = lane >> 4;

    v16h af[NK];
#pragma unroll
    for (int ks = 0; ks < NK; ++ks)
        af[ks] = load_wfrag<CIN>(W, cout0, ks * 32, lane);

    v16h sc[NK], sh[NK];
    if (!FIRST) {
#pragma unroll
        for (int ks = 0; ks < NK; ++ks) {
            const _Float16* sp = scale + ks * 32 + g * 16;
            const _Float16* tp = shift + ks * 32 + g * 16;
            v8h s0 = *(const v8h*)sp, s1 = *(const v8h*)(sp + 8);
            v8h t0 = *(const v8h*)tp, t1 = *(const v8h*)(tp + 8);
#pragma unroll
            for (int h = 0; h < 8; ++h) {
                sc[ks][h] = s0[h]; sc[ks][h + 8] = s1[h];
                sh[ks][h] = t0[h]; sh[ks][h + 8] = t1[h];
            }
        }
    }

    float ssum[8], ssq[8];
#pragma unroll
    for (int i = 0; i < 8; ++i) { ssum[i] = 0.0f; ssq[i] = 0.0f; }

    const int stride = gridDim.x * MPB;
    for (int mt = blockIdx.x * MPB + msub; mt < NTILES; mt += stride) {
        const int m = mt * 16 + (lane & 15);
        v8f acc;
#pragma unroll
        for (int i = 0; i < 8; ++i) acc[i] = 0.0f;

#pragma unroll
        for (int ks = 0; ks < NK; ++ks) {
            v16h bb;
            if (FIRST) {
#pragma unroll
                for (int h = 0; h < 16; ++h) bb[h] = (_Float16)0;
                if (lane < 16) {
                    v8h r = *(const v8h*)(Ain + (size_t)m * CS);
#pragma unroll
                    for (int h = 0; h < 8; ++h) bb[h] = r[h];
                }
            } else {
                const _Float16* p = Ain + (size_t)m * CS + ks * 32 + g * 16;
                __builtin_prefetch(p + (size_t)stride * 16 * CS, 0, 1);
                v8h r0 = *(const v8h*)p;
                v8h r1 = *(const v8h*)(p + 8);
#pragma unroll
                for (int h = 0; h < 8; ++h) { bb[h] = r0[h]; bb[h + 8] = r1[h]; }
                bb = bb * sc[ks] + sh[ks];          // per-channel BN affine
#pragma unroll
                for (int h = 0; h < 16; ++h)        // relu
                    if (bb[h] < (_Float16)0) bb[h] = (_Float16)0;
            }
            acc = __builtin_amdgcn_wmma_f32_16x16x32_f16(
                false, af[ks], false, bb, (short)0, acc, false, false);
        }

        v8h o;
#pragma unroll
        for (int i = 0; i < 8; ++i) {
            float v = acc[i];
            ssum[i] += v; ssq[i] += v * v;
            o[i] = (_Float16)v;
        }
        *(v8h*)(Yout + (size_t)m * COUT + cout0 + g * 8) = o;
    }

    // reduce stats across the 16 lanes of each half-wave, then atomics
#pragma unroll
    for (int off = 1; off < 16; off <<= 1) {
#pragma unroll
        for (int i = 0; i < 8; ++i) {
            ssum[i] += __shfl_xor(ssum[i], off, 32);
            ssq[i]  += __shfl_xor(ssq[i],  off, 32);
        }
    }
    if ((lane & 15) == 0) {
#pragma unroll
        for (int i = 0; i < 8; ++i) {
            int c = cout0 + g * 8 + i;
            atomicAdd(&gsum[c], ssum[i]);
            atomicAdd(&gsq[c],  ssq[i]);
        }
    }
}

// ---------------- BN finalize: (sum,sumsq,g,beta) -> f16 (scale,shift) ----
__global__ void finalize_kernel(const float* __restrict__ sum,
                                const float* __restrict__ sq,
                                const float* __restrict__ gam,
                                const float* __restrict__ beta,
                                _Float16* __restrict__ scale,
                                _Float16* __restrict__ shift, int C) {
    int c = threadIdx.x;
    if (c >= C) return;
    const float invM = 1.0f / (float)MTOT;
    float mean = sum[c] * invM;
    float var  = sq[c] * invM - mean * mean;
    float s    = gam[c] * rsqrtf(var + 1e-5f);
    scale[c] = (_Float16)s;
    shift[c] = (_Float16)(beta[c] - mean * s);
}

// ---------------- norm+relu+max over K ----------------
__global__ void maxpool_kernel(const _Float16* __restrict__ Y2,
                               const _Float16* __restrict__ scale,
                               const _Float16* __restrict__ shift,
                               float* __restrict__ out_feat) {
    const int gw   = blockIdx.x * 8 + (threadIdx.x >> 5);
    const int lane = threadIdx.x & 31;
    const int b = gw >> 10, s = gw & 1023;
    const int c0 = lane * 4;
    float sc[4], sh[4];
#pragma unroll
    for (int i = 0; i < 4; ++i) { sc[i] = (float)scale[c0 + i]; sh[i] = (float)shift[c0 + i]; }
    float acc[4];
#pragma unroll
    for (int i = 0; i < 4; ++i) acc[i] = -1e30f;
    const _Float16* yp = Y2 + (size_t)(b * NGRP + s) * KGRP * 128 + c0;
#pragma unroll 4
    for (int m = 0; m < KGRP; ++m) {
        v4h r = *(const v4h*)(yp + (size_t)m * 128);
#pragma unroll
        for (int i = 0; i < 4; ++i) {
            float v = (float)r[i] * sc[i] + sh[i];
            acc[i] = fmaxf(acc[i], v);
        }
    }
#pragma unroll
    for (int i = 0; i < 4; ++i)
        out_feat[((size_t)b * 128 + c0 + i) * NGRP + s] = fmaxf(acc[i], 0.0f);
}

__global__ void zero_kernel(float* __restrict__ p, int n) {
    int i = blockIdx.x * blockDim.x + threadIdx.x;
    if (i < n) p[i] = 0.0f;
}

// ---------------- launch ----------------
extern "C" void kernel_launch(void* const* d_in, const int* in_sizes, int n_in,
                              void* d_out, int out_size, void* d_ws, size_t ws_size,
                              hipStream_t stream) {
    const float* feature = (const float*)d_in[0];
    const float* coord   = (const float*)d_in[1];
    const float* W0 = (const float*)d_in[2];
    const float* g0 = (const float*)d_in[4];
    const float* be0 = (const float*)d_in[5];
    const float* W1 = (const float*)d_in[6];
    const float* g1 = (const float*)d_in[8];
    const float* be1 = (const float*)d_in[9];
    const float* W2 = (const float*)d_in[10];
    const float* g2 = (const float*)d_in[12];
    const float* be2 = (const float*)d_in[13];

    // workspace layout (bytes)
    const size_t O_NXYZ   = 0;                               // 196608 B
    const size_t O_STATS  = 196608;                          // 512 floats
    const size_t O_SCALES = 198656;                          // 512 halves
    const size_t O_Y1     = 262144;                          // 134 MB
    const size_t O_A0     = O_Y1 + (size_t)134217728;        // 16 MB
    const size_t O_Y0     = O_A0 + (size_t)16777216;         // 134 MB
    const size_t O_Y2     = O_A0;                            // overlays A0+Y0 (dead), 268 MB

    char* ws = (char*)d_ws;
    float*    nxyz   = (float*)(ws + O_NXYZ);
    float*    stats  = (float*)(ws + O_STATS);
    _Float16* scales = (_Float16*)(ws + O_SCALES);
    _Float16* Y1 = (_Float16*)(ws + O_Y1);
    _Float16* A0 = (_Float16*)(ws + O_A0);
    _Float16* Y0 = (_Float16*)(ws + O_Y0);
    _Float16* Y2 = (_Float16*)(ws + O_Y2);

    float* out_feat  = (float*)d_out;
    float* out_coord = out_feat + (size_t)NBATCH * 128 * NGRP;

    zero_kernel<<<2, 256, 0, stream>>>(stats, 512);
    fps_kernel<<<NBATCH, 256, 0, stream>>>(coord, nxyz, out_coord);
    group_kernel<<<2048, 256, 0, stream>>>(coord, feature, nxyz, A0);

    gemm_kernel<6, 8, 64, true><<<2048, 256, 0, stream>>>(
        A0, W0, (const _Float16*)nullptr, (const _Float16*)nullptr,
        Y0, stats + 0, stats + 64);
    finalize_kernel<<<1, 64, 0, stream>>>(stats + 0, stats + 64, g0, be0,
                                          scales + 0, scales + 64, 64);
    gemm_kernel<64, 64, 64, false><<<2048, 256, 0, stream>>>(
        Y0, W1, scales + 0, scales + 64, Y1, stats + 128, stats + 192);
    finalize_kernel<<<1, 64, 0, stream>>>(stats + 128, stats + 192, g1, be1,
                                          scales + 128, scales + 192, 64);
    gemm_kernel<64, 64, 128, false><<<2048, 256, 0, stream>>>(
        Y1, W2, scales + 128, scales + 192, Y2, stats + 256, stats + 384);
    finalize_kernel<<<1, 128, 0, stream>>>(stats + 256, stats + 384, g2, be2,
                                           scales + 256, scales + 384, 128);
    maxpool_kernel<<<2048, 256, 0, stream>>>(Y2, scales + 256, scales + 384, out_feat);
}
